// Net_6416681140530
// MI455X (gfx1250) — matrix operand order repaired and checked by reading, not compiled, not run on hardware
//
#include <hip/hip_runtime.h>
#include <hip/hip_bf16.h>
#include <math.h>

// ---------------------------------------------------------------------------
// MPNN (NNConv + GRU + Set2Set) for MI455X (gfx1250, wave32, WMMA).
// Heavy op: fused edge GEMM msg[e,:] using v_wmma_f32_16x16x32_bf16 with an
// L2-resident 1MB bf16 copy of nn_w2; w_e (655MB) is never materialized.
// ---------------------------------------------------------------------------

typedef __bf16 bf16;
typedef __attribute__((ext_vector_type(16))) __bf16 v16bf;
typedef __attribute__((ext_vector_type(8)))  __bf16 v8bf;
typedef __attribute__((ext_vector_type(8)))  float  v8f;

#define NDIM 64

__device__ __forceinline__ float sigmoidf_(float x) { return 1.0f / (1.0f + expf(-x)); }
__device__ __forceinline__ unsigned fenc(float x) {
  unsigned u = __float_as_uint(x);
  return (u & 0x80000000u) ? ~u : (u | 0x80000000u);
}
__device__ __forceinline__ float fdec(unsigned u) {
  return (u & 0x80000000u) ? __uint_as_float(u & 0x7fffffffu) : __uint_as_float(~u);
}
#define ENC_NEG_INF 0x007FFFFFu  // fenc(-inf)

// ------------------------------ small kernels ------------------------------

__global__ void k_zero(float* __restrict__ p, int n) {
  int i = blockIdx.x * blockDim.x + threadIdx.x;
  if (i < n) p[i] = 0.0f;
}

__global__ void k_deg(const int* __restrict__ dst, float* __restrict__ deg, int E) {
  int e = blockIdx.x * blockDim.x + threadIdx.x;
  if (e < E) atomicAdd(&deg[dst[e]], 1.0f);
}

__global__ void k_invdeg(const float* __restrict__ deg, float* __restrict__ inv, int N) {
  int n = blockIdx.x * blockDim.x + threadIdx.x;
  if (n < N) inv[n] = 1.0f / fmaxf(deg[n], 1.0f);
}

// out = relu(x @ lin0_w.T + b); one block per node, 64 threads
__global__ void __launch_bounds__(64) k_lin0(const float* __restrict__ x,
                                             const float* __restrict__ w,
                                             const float* __restrict__ b,
                                             float* __restrict__ out_f,
                                             bf16* __restrict__ out_b, int N) {
  int n = blockIdx.x, o = threadIdx.x;
  const float* xr = x + (size_t)n * 21;
  const float* wr = w + (size_t)o * 21;
  float acc = b[o];
#pragma unroll
  for (int j = 0; j < 21; ++j) acc += xr[j] * wr[j];
  acc = fmaxf(acc, 0.0f);
  out_f[(size_t)n * NDIM + o] = acc;
  out_b[(size_t)n * NDIM + o] = (bf16)acc;
}

// e_h = relu(edge_attr @ nn_w1.T + b1) stored bf16 [E,128]
__global__ void __launch_bounds__(128) k_edgenet(const float* __restrict__ ea,
                                                 const float* __restrict__ w1,
                                                 const float* __restrict__ b1,
                                                 bf16* __restrict__ eh, int E) {
  int e = blockIdx.x, c = threadIdx.x;
  const float* er = ea + (size_t)e * 16;
  const float* wr = w1 + (size_t)c * 16;
  float acc = b1[c];
#pragma unroll
  for (int j = 0; j < 16; ++j) acc += er[j] * wr[j];
  eh[(size_t)e * 128 + c] = (bf16)fmaxf(acc, 0.0f);
}

// Pack nn_w2 [4096,128] f32 into per-lane WMMA B fragments, bf16.
// Linear layout: idx = ((c*4 + nt)*32 + L)*16 + t  (c=K-chunk 0..255, nt=N tile,
// L=lane, t=element). B[k_local, n] pattern: lanes 0-15 hold K {0-7,16-23}, lanes
// 16-31 hold K {8-15,24-31} of the 32-wide chunk; n = nt*16 + (L&15).
__global__ void k_prep_b(const float* __restrict__ nn_w2, bf16* __restrict__ Bfrag) {
  int idx = blockIdx.x * blockDim.x + threadIdx.x;  // 0..524287
  int t = idx & 15;
  int L = (idx >> 4) & 31;
  int nt = (idx >> 9) & 3;
  int c = idx >> 11;
  int kloc = ((L & 16) ? 8 : 0) + (t & 7) + ((t & 8) ? 16 : 0);
  int j = c * 32 + kloc;  // flat K index 0..8191 : j = i*128 + k
  int i = j >> 7;
  int k = j & 127;
  int o = nt * 16 + (L & 15);
  Bfrag[idx] = (bf16)nn_w2[(size_t)(i * NDIM + o) * 128 + k];
}

// ----------------------- fused edge WMMA message kernel --------------------
// msg[e,o] = sum_i out[src[e],i] * (sum_k e_h[e,k] * W2[i*64+o, k])
// block = 256 threads = 8 waves: 2 edge-tiles(16) x 4 n-tiles(16). K=128 per i.
// Scatters raw segment sums into agg (and out[src] sums into Ssum for nn_b2).
__global__ void __launch_bounds__(256) k_msg_wmma(const bf16* __restrict__ eh,
                                                  const bf16* __restrict__ outb,
                                                  const bf16* __restrict__ Bfrag,
                                                  const int* __restrict__ src,
                                                  const int* __restrict__ dst,
                                                  float* __restrict__ agg,
                                                  float* __restrict__ Ssum, int E) {
  __shared__ __align__(16) bf16 lds_out[32][NDIM];  // out[src[e],:] for 32 edges
  const int tid = threadIdx.x;
  const int lane = tid & 31;
  const int wave = tid >> 5;
  const int etile = wave >> 2;  // 0..1
  const int nt = wave & 3;      // 0..3
  const int e0 = blockIdx.x * 32;

  {  // cooperative stage of source-node features (zero rows for tail edges)
    int row = tid >> 3;
    int col = (tid & 7) * 8;
    int e = e0 + row;
    if (e < E) {
      int s = src[e];
      *(uint4*)(&lds_out[row][col]) = *(const uint4*)(outb + (size_t)s * NDIM + col);
    } else {
      uint4 z = {0u, 0u, 0u, 0u};
      *(uint4*)(&lds_out[row][col]) = z;
    }
  }
  __syncthreads();

  // A fragments: 16 edge rows x 32-K chunks of e_h (4 chunks -> K=0..127).
  // 16-bit A layout: lanes 0-15 hold K {0-7,16-23}; lanes 16-31 K {8-15,24-31}.
  int e_a = e0 + etile * 16 + (lane & 15);
  if (e_a >= E) e_a = 0;  // safe: those D rows are scaled by 0 and never stored
  const bf16* ehrow = eh + (size_t)e_a * 128;
  const int lo = (lane & 16) ? 8 : 0;
  v16bf ef0, ef1, ef2, ef3;
#define LOAD_EF(dstv, kb)                                                      \
  {                                                                            \
    const bf16* q = ehrow + (kb) + lo;                                         \
    v8bf a0 = *(const v8bf*)(q);                                               \
    v8bf a1 = *(const v8bf*)(q + 16);                                          \
    dstv = __builtin_shufflevector(a0, a1, 0, 1, 2, 3, 4, 5, 6, 7, 8, 9, 10,   \
                                   11, 12, 13, 14, 15);                        \
  }
  LOAD_EF(ef0, 0)
  LOAD_EF(ef1, 32)
  LOAD_EF(ef2, 64)
  LOAD_EF(ef3, 96)
#undef LOAD_EF

  v8f acc = {0.f, 0.f, 0.f, 0.f, 0.f, 0.f, 0.f, 0.f};
  const int mbase = etile * 16 + ((lane & 16) ? 8 : 0);  // D-tile row base

  for (int i = 0; i < NDIM; ++i) {
    const bf16* bp = Bfrag + (size_t)i * 8192 + (size_t)nt * 512 + (size_t)lane * 16;
    v16bf b0 = *(const v16bf*)(bp);
    v16bf b1 = *(const v16bf*)(bp + 2048);
    v16bf b2 = *(const v16bf*)(bp + 4096);
    v16bf b3 = *(const v16bf*)(bp + 6144);
    v8f t = {0.f, 0.f, 0.f, 0.f, 0.f, 0.f, 0.f, 0.f};
    t = __builtin_amdgcn_wmma_f32_16x16x32_bf16(false, ef0, false, b0, (short)0, t, false, false);
    t = __builtin_amdgcn_wmma_f32_16x16x32_bf16(false, ef1, false, b1, (short)0, t, false, false);
    t = __builtin_amdgcn_wmma_f32_16x16x32_bf16(false, ef2, false, b2, (short)0, t, false, false);
    t = __builtin_amdgcn_wmma_f32_16x16x32_bf16(false, ef3, false, b3, (short)0, t, false, false);
#pragma unroll
    for (int r = 0; r < 8; ++r) {
      float sf = (float)lds_out[mbase + r][i];  // out[src[e_row], i], bank-bcast
      acc[r] += sf * t[r];
    }
  }

  // scatter: agg[dst[e],o] += msg ; Ssum[dst[e],o] += out[src[e],o]
  const int ncol = nt * 16 + (lane & 15);
#pragma unroll
  for (int r = 0; r < 8; ++r) {
    int e_m = e0 + mbase + r;
    if (e_m < E) {
      int d = dst[e_m];
      atomicAdd(&agg[(size_t)d * NDIM + ncol], acc[r]);
      atomicAdd(&Ssum[(size_t)d * NDIM + ncol], (float)lds_out[mbase + r][ncol]);
    }
  }
}

// NNConv root/bias + GRU cell. One block per node, 64 threads.
__global__ void __launch_bounds__(64) k_node_update(
    const float* __restrict__ agg, const float* __restrict__ Ssum,
    const float* __restrict__ inv_deg, float* __restrict__ out_f,
    bf16* __restrict__ out_b, const float* __restrict__ conv_root,
    const float* __restrict__ conv_bias, const float* __restrict__ nn_b2,
    const float* __restrict__ w_ih, const float* __restrict__ w_hh,
    const float* __restrict__ b_ih, const float* __restrict__ b_hh, int N) {
  int n = blockIdx.x, o = threadIdx.x;
  __shared__ float lm[NDIM], lh[NDIM], lS[NDIM];
  float h_o = out_f[(size_t)n * NDIM + o];
  lh[o] = h_o;
  lS[o] = Ssum[(size_t)n * NDIM + o];
  __syncthreads();
  float root = 0.f, bias2 = 0.f;
#pragma unroll 8
  for (int i = 0; i < NDIM; ++i) {
    root += lh[i] * conv_root[i * NDIM + o];
    bias2 += lS[i] * nn_b2[i * NDIM + o];
  }
  float aggv = (agg[(size_t)n * NDIM + o] + bias2) * inv_deg[n];
  float m = fmaxf(aggv + root + conv_bias[o], 0.0f);
  lm[o] = m;
  __syncthreads();
  float gir = b_ih[o], giz = b_ih[NDIM + o], gin = b_ih[2 * NDIM + o];
  float ghr = b_hh[o], ghz = b_hh[NDIM + o], ghn = b_hh[2 * NDIM + o];
#pragma unroll 8
  for (int i = 0; i < NDIM; ++i) {
    float mi = lm[i], hi = lh[i];
    gir += mi * w_ih[o * NDIM + i];
    giz += mi * w_ih[(NDIM + o) * NDIM + i];
    gin += mi * w_ih[(2 * NDIM + o) * NDIM + i];
    ghr += hi * w_hh[o * NDIM + i];
    ghz += hi * w_hh[(NDIM + o) * NDIM + i];
    ghn += hi * w_hh[(2 * NDIM + o) * NDIM + i];
  }
  float r = sigmoidf_(gir + ghr);
  float z = sigmoidf_(giz + ghz);
  float nn = tanhf(gin + r * ghn);
  float hn = (1.0f - z) * nn + z * h_o;
  out_f[(size_t)n * NDIM + o] = hn;
  out_b[(size_t)n * NDIM + o] = (bf16)hn;
}

// ------------------------------ Set2Set ------------------------------------

__global__ void __launch_bounds__(128) k_s2s_init(float* __restrict__ qstar,
                                                  float* __restrict__ hs,
                                                  float* __restrict__ cs) {
  int b = blockIdx.x, j = threadIdx.x;
  qstar[(size_t)b * 128 + j] = 0.f;
  if (j < NDIM) { hs[(size_t)b * NDIM + j] = 0.f; cs[(size_t)b * NDIM + j] = 0.f; }
}

__global__ void __launch_bounds__(64) k_lstm(const float* __restrict__ qstar,
                                             float* __restrict__ hs, float* __restrict__ cs,
                                             const float* __restrict__ w_ih,
                                             const float* __restrict__ w_hh,
                                             const float* __restrict__ b_ih,
                                             const float* __restrict__ b_hh) {
  int b = blockIdx.x, o = threadIdx.x;
  __shared__ float lq[128], lh[NDIM];
  lq[o] = qstar[(size_t)b * 128 + o];
  lq[NDIM + o] = qstar[(size_t)b * 128 + NDIM + o];
  lh[o] = hs[(size_t)b * NDIM + o];
  __syncthreads();
  float g[4];
#pragma unroll
  for (int gg = 0; gg < 4; ++gg) {
    int idx = gg * NDIM + o;
    float acc = b_ih[idx] + b_hh[idx];
    for (int j = 0; j < 128; ++j) acc += lq[j] * w_ih[(size_t)idx * 128 + j];
    for (int j = 0; j < NDIM; ++j) acc += lh[j] * w_hh[(size_t)idx * NDIM + j];
    g[gg] = acc;
  }
  float c_new = sigmoidf_(g[1]) * cs[(size_t)b * NDIM + o] + sigmoidf_(g[0]) * tanhf(g[2]);
  cs[(size_t)b * NDIM + o] = c_new;
  hs[(size_t)b * NDIM + o] = sigmoidf_(g[3]) * tanhf(c_new);
}

__global__ void __launch_bounds__(64) k_attn_init(float* __restrict__ r_read,
                                                  float* __restrict__ s_sum,
                                                  unsigned* __restrict__ mmax_u) {
  int b = blockIdx.x, o = threadIdx.x;
  r_read[(size_t)b * NDIM + o] = 0.f;
  if (o == 0) { s_sum[b] = 0.f; mmax_u[b] = ENC_NEG_INF; }
}

__global__ void __launch_bounds__(64) k_score(const float* __restrict__ out_f,
                                              const float* __restrict__ hs,
                                              const int* __restrict__ batch,
                                              float* __restrict__ e_sc,
                                              unsigned* __restrict__ mmax_u, int N) {
  int n = blockIdx.x, o = threadIdx.x;
  int bb = batch[n];
  __shared__ float red[NDIM];
  red[o] = out_f[(size_t)n * NDIM + o] * hs[(size_t)bb * NDIM + o];
  __syncthreads();
  for (int st = 32; st > 0; st >>= 1) {
    if (o < st) red[o] += red[o + st];
    __syncthreads();
  }
  if (o == 0) {
    e_sc[n] = red[0];
    atomicMax(&mmax_u[bb], fenc(red[0]));
  }
}

__global__ void k_expsum(const float* __restrict__ e_sc, const unsigned* __restrict__ mmax_u,
                         const int* __restrict__ batch, float* __restrict__ ex,
                         float* __restrict__ s_sum, int N) {
  int n = blockIdx.x * blockDim.x + threadIdx.x;
  if (n < N) {
    int bb = batch[n];
    float v = expf(e_sc[n] - fdec(mmax_u[bb]));
    ex[n] = v;
    atomicAdd(&s_sum[bb], v);
  }
}

__global__ void __launch_bounds__(64) k_read(const float* __restrict__ out_f,
                                             const float* __restrict__ ex,
                                             const float* __restrict__ s_sum,
                                             const int* __restrict__ batch,
                                             float* __restrict__ r_read, int N) {
  int n = blockIdx.x, o = threadIdx.x;
  int bb = batch[n];
  float a = ex[n] / s_sum[bb];
  atomicAdd(&r_read[(size_t)bb * NDIM + o], a * out_f[(size_t)n * NDIM + o]);
}

__global__ void __launch_bounds__(128) k_qstar(const float* __restrict__ hs,
                                               const float* __restrict__ r_read,
                                               float* __restrict__ qstar) {
  int b = blockIdx.x, j = threadIdx.x;
  qstar[(size_t)b * 128 + j] =
      (j < NDIM) ? hs[(size_t)b * NDIM + j] : r_read[(size_t)b * NDIM + (j - NDIM)];
}

__global__ void __launch_bounds__(64) k_final(const float* __restrict__ qstar,
                                              const float* __restrict__ l1w,
                                              const float* __restrict__ l1b,
                                              const float* __restrict__ l2w,
                                              const float* __restrict__ l2b,
                                              float* __restrict__ y) {
  int b = blockIdx.x, o = threadIdx.x;
  float acc = l1b[o];
  for (int j = 0; j < 128; ++j) acc += qstar[(size_t)b * 128 + j] * l1w[(size_t)o * 128 + j];
  __shared__ float red[NDIM];
  red[o] = fmaxf(acc, 0.0f) * l2w[o];
  __syncthreads();
  for (int st = 32; st > 0; st >>= 1) {
    if (o < st) red[o] += red[o + st];
    __syncthreads();
  }
  if (o == 0) y[b] = red[0] + l2b[0];
}

// ------------------------------ launcher -----------------------------------

extern "C" void kernel_launch(void* const* d_in, const int* in_sizes, int n_in,
                              void* d_out, int out_size, void* d_ws, size_t ws_size,
                              hipStream_t stream) {
  (void)n_in; (void)ws_size;
  const float* x         = (const float*)d_in[0];
  const float* edge_attr = (const float*)d_in[1];
  const int*   edge_idx  = (const int*)d_in[2];
  const int*   batch     = (const int*)d_in[3];
  const float* lin0_w    = (const float*)d_in[4];
  const float* lin0_b    = (const float*)d_in[5];
  const float* nn_w1     = (const float*)d_in[6];
  const float* nn_b1     = (const float*)d_in[7];
  const float* nn_w2     = (const float*)d_in[8];
  const float* nn_b2     = (const float*)d_in[9];
  const float* conv_root = (const float*)d_in[10];
  const float* conv_bias = (const float*)d_in[11];
  const float* gru_w_ih  = (const float*)d_in[12];
  const float* gru_w_hh  = (const float*)d_in[13];
  const float* gru_b_ih  = (const float*)d_in[14];
  const float* gru_b_hh  = (const float*)d_in[15];
  const float* lstm_w_ih = (const float*)d_in[16];
  const float* lstm_w_hh = (const float*)d_in[17];
  const float* lstm_b_ih = (const float*)d_in[18];
  const float* lstm_b_hh = (const float*)d_in[19];
  const float* lin1_w    = (const float*)d_in[20];
  const float* lin1_b    = (const float*)d_in[21];
  const float* lin2_w    = (const float*)d_in[22];
  const float* lin2_b    = (const float*)d_in[23];

  const int N = in_sizes[0] / 21;
  const int E = in_sizes[1] / 16;
  const int B = out_size;  // NUM_GRAPHS
  const int* src = edge_idx;
  const int* dst = edge_idx + E;

  char* ws = (char*)d_ws;
  size_t off = 0;
  auto alloc = [&](size_t bytes) -> void* {
    size_t o = (off + 255) & ~(size_t)255;
    off = o + bytes;
    return (void*)(ws + o);
  };
  float* out_f   = (float*)alloc((size_t)N * NDIM * 4);
  bf16*  out_b   = (bf16*)alloc((size_t)N * NDIM * 2);
  bf16*  eh      = (bf16*)alloc((size_t)E * 128 * 2);
  bf16*  Bfrag   = (bf16*)alloc((size_t)8192 * NDIM * 2);
  float* agg     = (float*)alloc((size_t)N * NDIM * 4);
  float* Ssum    = (float*)alloc((size_t)N * NDIM * 4);
  float* deg     = (float*)alloc((size_t)N * 4);
  float* inv_deg = (float*)alloc((size_t)N * 4);
  float* hs      = (float*)alloc((size_t)B * NDIM * 4);
  float* cs      = (float*)alloc((size_t)B * NDIM * 4);
  float* qstar   = (float*)alloc((size_t)B * 128 * 4);
  float* e_sc    = (float*)alloc((size_t)N * 4);
  float* ex      = (float*)alloc((size_t)N * 4);
  float* s_sum   = (float*)alloc((size_t)B * 4);
  unsigned* mmax = (unsigned*)alloc((size_t)B * 4);
  float* r_read  = (float*)alloc((size_t)B * NDIM * 4);

  // ---- encoders + constants ----
  k_lin0<<<N, 64, 0, stream>>>(x, lin0_w, lin0_b, out_f, out_b, N);
  k_edgenet<<<E, 128, 0, stream>>>(edge_attr, nn_w1, nn_b1, eh, E);
  k_prep_b<<<(8192 * NDIM) / 256, 256, 0, stream>>>(nn_w2, Bfrag);
  k_zero<<<(N + 255) / 256, 256, 0, stream>>>(deg, N);
  k_deg<<<(E + 255) / 256, 256, 0, stream>>>(dst, deg, E);
  k_invdeg<<<(N + 255) / 256, 256, 0, stream>>>(deg, inv_deg, N);

  // ---- 3 NNConv + GRU iterations ----
  for (int it = 0; it < 3; ++it) {
    k_zero<<<((N * NDIM) + 255) / 256, 256, 0, stream>>>(agg, N * NDIM);
    k_zero<<<((N * NDIM) + 255) / 256, 256, 0, stream>>>(Ssum, N * NDIM);
    k_msg_wmma<<<(E + 31) / 32, 256, 0, stream>>>(eh, out_b, Bfrag, src, dst, agg, Ssum, E);
    k_node_update<<<N, 64, 0, stream>>>(agg, Ssum, inv_deg, out_f, out_b, conv_root,
                                        conv_bias, nn_b2, gru_w_ih, gru_w_hh,
                                        gru_b_ih, gru_b_hh, N);
  }

  // ---- Set2Set pooling, 3 steps ----
  k_s2s_init<<<B, 128, 0, stream>>>(qstar, hs, cs);
  for (int st = 0; st < 3; ++st) {
    k_lstm<<<B, 64, 0, stream>>>(qstar, hs, cs, lstm_w_ih, lstm_w_hh, lstm_b_ih, lstm_b_hh);
    k_attn_init<<<B, 64, 0, stream>>>(r_read, s_sum, mmax);
    k_score<<<N, 64, 0, stream>>>(out_f, hs, batch, e_sc, mmax, N);
    k_expsum<<<(N + 255) / 256, 256, 0, stream>>>(e_sc, mmax, batch, ex, s_sum, N);
    k_read<<<N, 64, 0, stream>>>(out_f, ex, s_sum, batch, r_read, N);
    k_qstar<<<B, 128, 0, stream>>>(hs, r_read, qstar);
  }

  // ---- readout MLP ----
  k_final<<<B, 64, 0, stream>>>(qstar, lin1_w, lin1_b, lin2_w, lin2_b, (float*)d_out);
}
